// BlockWithCompression_41858751267282
// MI455X (gfx1250) — matrix operand order repaired and checked by reading, not compiled, not run on hardware
//
#include <hip/hip_runtime.h>

// ---------------------------------------------------------------------------
// Types
// ---------------------------------------------------------------------------
typedef __attribute__((ext_vector_type(16))) __bf16        v16bf;
typedef __attribute__((ext_vector_type(8)))  float         v8f;
typedef __attribute__((ext_vector_type(4))) unsigned int   u32x4;
typedef __attribute__((ext_vector_type(2))) unsigned int   u32x2;
typedef unsigned short us;

union FragU { u32x4 q[2]; v16bf v; };        // 32B: one 16-bit WMMA A/B fragment

__device__ inline us f2bf(float f) {          // round-to-nearest-even f32 -> bf16
  unsigned u = __float_as_uint(f);
  u += 0x7FFFu + ((u >> 16) & 1u);
  return (us)(u >> 16);
}

__device__ inline v8f zero8() { v8f z = {0.f,0.f,0.f,0.f,0.f,0.f,0.f,0.f}; return z; }

__device__ inline v8f wmma_bf16(const FragU& a, const FragU& b, v8f c) {
  return __builtin_amdgcn_wmma_f32_16x16x32_bf16(false, a.v, false, b.v,
                                                 (short)0, c, false, false);
}

// ---------------------------------------------------------------------------
// Async global->LDS copy (CDNA5 GLOBAL_LOAD_ASYNC_TO_LDS_B128, ASYNCcnt).
// Probe-derived prototype: (int4 global*, int4 shared*, imm offset, imm cpol).
// ---------------------------------------------------------------------------
#if defined(__has_builtin)
#  if __has_builtin(__builtin_amdgcn_global_load_async_to_lds_b128) && \
      __has_builtin(__builtin_amdgcn_s_wait_asynccnt)
#    define USE_ASYNC_COPY 1
#  endif
#endif
#ifndef USE_ASYNC_COPY
#  define USE_ASYNC_COPY 0
#endif

typedef __attribute__((__vector_size__(4 * sizeof(int)))) int vi4;
typedef __attribute__((address_space(1))) vi4* g_vi4_p;
typedef __attribute__((address_space(3))) vi4* l_vi4_p;

__device__ inline void cp_b128(us* dst, const us* src) {
#if USE_ASYNC_COPY
  __builtin_amdgcn_global_load_async_to_lds_b128((g_vi4_p)src, (l_vi4_p)dst, 0, 0);
#else
  *(u32x4*)dst = *(const u32x4*)src;
#endif
}

__device__ inline void wait_async() {
#if USE_ASYNC_COPY
  __builtin_amdgcn_s_wait_asynccnt(0);
#endif
}

// ---------------------------------------------------------------------------
// fp32 -> bf16 convert + transpose: W[K][N] -> WT[N][K]   (one-time per launch)
// Block handles a 32(K) x 32(N) tile through LDS so both sides stay coalesced.
// ---------------------------------------------------------------------------
__global__ __launch_bounds__(256) void k_cvtT(const float* __restrict__ W,
                                              us* __restrict__ WT, int K, int N) {
  __shared__ us ldsT[32 * 36];            // [n][k], stride 36 (8B-aligned rows)
  const int tid = threadIdx.x;
  const int k0 = blockIdx.x * 32, n0 = blockIdx.y * 32;
  {
    int kr = tid >> 3, nc4 = (tid & 7) * 4;
    float4 v = *(const float4*)(W + (size_t)(k0 + kr) * N + n0 + nc4);
    ldsT[(nc4 + 0) * 36 + kr] = f2bf(v.x);
    ldsT[(nc4 + 1) * 36 + kr] = f2bf(v.y);
    ldsT[(nc4 + 2) * 36 + kr] = f2bf(v.z);
    ldsT[(nc4 + 3) * 36 + kr] = f2bf(v.w);
  }
  __syncthreads();
  {
    int n = tid >> 3, kc4 = (tid & 7) * 4;
    *(u32x2*)(WT + (size_t)(n0 + n) * K + k0 + kc4) =
        *(const u32x2*)&ldsT[n * 36 + kc4];
  }
}

// ---------------------------------------------------------------------------
// LayerNorm over E=1024, one row per block (256 threads = 8 waves, wave32)
// ---------------------------------------------------------------------------
__global__ __launch_bounds__(256) void k_layernorm(const float* __restrict__ x,
                                                   const float* __restrict__ g,
                                                   const float* __restrict__ beta,
                                                   us* __restrict__ out) {
  const int row = blockIdx.x;
  const int tid = threadIdx.x;
  const float4 v = *(const float4*)(x + (size_t)row * 1024 + tid * 4);
  float s  = v.x + v.y + v.z + v.w;
  float s2 = v.x*v.x + v.y*v.y + v.z*v.z + v.w*v.w;
#pragma unroll
  for (int off = 16; off >= 1; off >>= 1) {
    s  += __shfl_xor(s,  off);
    s2 += __shfl_xor(s2, off);
  }
  __shared__ float red[16];
  __shared__ float sMu, sR;
  const int w = tid >> 5;
  if ((tid & 31) == 0) { red[w] = s; red[8 + w] = s2; }
  __syncthreads();
  if (tid == 0) {
    float a = 0.f, b2 = 0.f;
#pragma unroll
    for (int i = 0; i < 8; ++i) { a += red[i]; b2 += red[8 + i]; }
    float mu  = a * (1.0f / 1024.0f);
    float var = b2 * (1.0f / 1024.0f) - mu * mu;
    sMu = mu; sR = rsqrtf(var + 1e-5f);
  }
  __syncthreads();
  const float mu = sMu, r = sR;
  union { u32x2 d; us s4[4]; } p;
  const float vs[4] = {v.x, v.y, v.z, v.w};
#pragma unroll
  for (int j = 0; j < 4; ++j) {
    int col = tid * 4 + j;
    p.s4[j] = f2bf((vs[j] - mu) * r * g[col] + beta[col]);
  }
  *(u32x2*)(out + (size_t)row * 1024 + tid * 4) = p.d;
}

// ---------------------------------------------------------------------------
// bf16 GEMM: C[M,N] = A[M,K] @ BT[N,K]^T, f32 WMMA accumulate.
// 128x128 tile, 256 threads = 8 waves; wave = 32(M) x 64(N) = 2x4 frags.
// Double-buffered LDS; next K-tile copied (async if available) during compute.
// EPI 0: bf16 C   EPI 1: f32 res + C + bias   EPI 2: bf16 relu(C+bias)
// EPI 3: bf16 C scattered as V^T[b,h][d][t]
// ---------------------------------------------------------------------------
__device__ inline void stage_tile(us* dst, const us* srcBase, int k0, int K, int tid) {
#pragma unroll
  for (int j = 0; j < 2; ++j) {
    int c = tid + j * 256;
    int row = c >> 2, ch = c & 3;
    cp_b128(&dst[row * 40 + ch * 8], srcBase + (size_t)row * K + k0 + ch * 8);
  }
}

template <int EPI>
__global__ __launch_bounds__(256) void k_gemm(const us* __restrict__ A,
                                              const us* __restrict__ BT,
                                              const float* __restrict__ bias,
                                              const float* __restrict__ res,
                                              us* __restrict__ outB,
                                              float* __restrict__ outF,
                                              int M, int N, int K) {
  __shared__ us ldsA[2][128 * 40];   // [row m][k], stride 40 (16B-aligned rows)
  __shared__ us ldsB[2][128 * 40];   // [row n][k], stride 40
  const int tid = threadIdx.x;
  const int L = tid & 31, w = tid >> 5;
  const int waveM = w & 3, waveN = w >> 2;
  const int m0 = blockIdx.y * 128, n0 = blockIdx.x * 128;
  const us* Abase = A  + (size_t)m0 * K;
  const us* Bbase = BT + (size_t)n0 * K;

  v8f acc[2][4];
#pragma unroll
  for (int i = 0; i < 2; ++i)
#pragma unroll
    for (int j = 0; j < 4; ++j) acc[i][j] = zero8();

  stage_tile(ldsA[0], Abase, 0, K, tid);
  stage_tile(ldsB[0], Bbase, 0, K, tid);
  int p = 0;
  const int nk = K >> 5;
  for (int kt = 0; kt < nk; ++kt) {
    wait_async();
    __syncthreads();                       // buffer p loaded & visible block-wide
    if (kt + 1 < nk) {                     // overlap next-tile copy with WMMA
      stage_tile(ldsA[p ^ 1], Abase, (kt + 1) * 32, K, tid);
      stage_tile(ldsB[p ^ 1], Bbase, (kt + 1) * 32, K, tid);
    }
    const us* la = ldsA[p];
    const us* lb = ldsB[p];

    FragU bfr[4];
#pragma unroll
    for (int nf = 0; nf < 4; ++nf) {
      int n = waveN * 64 + nf * 16 + (L & 15);
      int base = n * 40 + (L >> 4) * 8;
      bfr[nf].q[0] = *(const u32x4*)&lb[base];
      bfr[nf].q[1] = *(const u32x4*)&lb[base + 16];
    }
#pragma unroll
    for (int mf = 0; mf < 2; ++mf) {
      FragU af;
      int rowA = waveM * 32 + mf * 16 + (L & 15);
      int abase = rowA * 40 + (L >> 4) * 8;
      af.q[0] = *(const u32x4*)&la[abase];
      af.q[1] = *(const u32x4*)&la[abase + 16];
#pragma unroll
      for (int nf = 0; nf < 4; ++nf)
        acc[mf][nf] = wmma_bf16(af, bfr[nf], acc[mf][nf]);
    }
    p ^= 1;
  }

  // epilogue (C/D layout: lane L holds N = L&15, M = (L>>4)*8 + i)
#pragma unroll
  for (int mf = 0; mf < 2; ++mf)
#pragma unroll
    for (int nf = 0; nf < 4; ++nf)
#pragma unroll
      for (int i = 0; i < 8; ++i) {
        int m = m0 + waveM * 32 + mf * 16 + ((L >> 4) << 3) + i;
        int n = n0 + waveN * 64 + nf * 16 + (L & 15);
        float vv = acc[mf][nf][i];
        if constexpr (EPI == 0) {
          outB[(size_t)m * N + n] = f2bf(vv);
        } else if constexpr (EPI == 1) {
          size_t idx = (size_t)m * N + n;
          outF[idx] = res[idx] + vv + bias[n];
        } else if constexpr (EPI == 2) {
          outB[(size_t)m * N + n] = f2bf(fmaxf(vv + bias[n], 0.0f));
        } else {  // EPI == 3: V^T layout [b*16+h][d][t], row length T=2048
          int bb = m >> 11, tt = m & 2047;
          size_t vidx = (((size_t)(bb * 16 + (n >> 6)) * 64 + (n & 63)) << 11) + tt;
          outB[vidx] = f2bf(vv);
        }
      }
}

// ---------------------------------------------------------------------------
// Causal flash attention. Grid (T/64, B*H). Block 128 = 4 waves.
// Wave = 16 queries; online softmax over 64-key tiles; S=QK^T and O+=PV via
// WMMA bf16 (scale = E^-0.5 = 1/32). K and V^T staged via (async) b128 copies
// into double-buffered LDS.
// ---------------------------------------------------------------------------
__device__ inline void stage_attn(us* dK, us* dVT, const us* kp, const us* vtp,
                                  int k0, int tid) {
  int r = tid >> 1;               // 0..63  (K: key row ; VT: d row)
  int c32 = (tid & 1) * 32;
#pragma unroll
  for (int j = 0; j < 4; ++j) {
    cp_b128(&dK[r * 72 + c32 + j * 8],  kp  + (size_t)(k0 + r) * 1024 + c32 + j * 8);
    cp_b128(&dVT[r * 72 + c32 + j * 8], vtp + (size_t)r * 2048 + k0 + c32 + j * 8);
  }
}

__global__ __launch_bounds__(128) void k_attention(const us* __restrict__ q,
                                                   const us* __restrict__ kmat,
                                                   const us* __restrict__ vt,
                                                   us* __restrict__ attn) {
  __shared__ us ldsK[2][64 * 72];      // [key][d]
  __shared__ us ldsVT[2][64 * 72];     // [d][key]
  __shared__ us ldsP[4][16 * 72];      // per-wave P relayout buffer
  const int tid = threadIdx.x;
  const int L = tid & 31, w = tid >> 5;
  const int bh = blockIdx.y, b = bh >> 4, h = bh & 15;
  const int q0 = blockIdx.x * 64;
  const int qw0 = q0 + w * 16;
  const size_t bo = ((size_t)b * 2048) * 1024 + h * 64;
  const us* qp = q + bo;
  const us* kp = kmat + bo;
  const us* vtp = vt + (size_t)bh * 64 * 2048;

  // Q A-fragments (reused across all key tiles): per-lane contiguous reads
  FragU qf[2];
  {
    int qrow = qw0 + (L & 15);
    int dblk = (L >> 4) * 8;
#pragma unroll
    for (int t = 0; t < 2; ++t) {
      qf[t].q[0] = *(const u32x4*)(qp + (size_t)qrow * 1024 + t * 32 + dblk);
      qf[t].q[1] = *(const u32x4*)(qp + (size_t)qrow * 1024 + t * 32 + 16 + dblk);
    }
  }

  v8f o[4];
#pragma unroll
  for (int i = 0; i < 4; ++i) o[i] = zero8();
  float ml[8], ll[8];
#pragma unroll
  for (int i = 0; i < 8; ++i) { ml[i] = -__builtin_inff(); ll[i] = 0.0f; }

  const int ntiles = blockIdx.x + 1;
  stage_attn(ldsK[0], ldsVT[0], kp, vtp, 0, tid);
  int p = 0;
  for (int kt = 0; kt < ntiles; ++kt) {
    const int k0 = kt * 64;
    wait_async();
    __syncthreads();
    if (kt + 1 < ntiles)
      stage_attn(ldsK[p ^ 1], ldsVT[p ^ 1], kp, vtp, (kt + 1) * 64, tid);

    if (k0 <= qw0 + 15) {                 // else fully masked for this wave
      const us* lk = ldsK[p];
      const us* lv = ldsVT[p];
      // S = Q K^T : 16 x 64 per wave (4 N-frags x 2 K-steps)
      v8f s[4];
#pragma unroll
      for (int nf = 0; nf < 4; ++nf) {
        FragU kf0, kf1;
        int krow = nf * 16 + (L & 15);
        int d0 = (L >> 4) * 8;
        kf0.q[0] = *(const u32x4*)&lk[krow * 72 + d0];
        kf0.q[1] = *(const u32x4*)&lk[krow * 72 + d0 + 16];
        kf1.q[0] = *(const u32x4*)&lk[krow * 72 + 32 + d0];
        kf1.q[1] = *(const u32x4*)&lk[krow * 72 + 32 + d0 + 16];
        v8f z = zero8();
        z = wmma_bf16(qf[0], kf0, z);
        z = wmma_bf16(qf[1], kf1, z);
        s[nf] = z;
      }

      const bool domask = (k0 + 63) > qw0;
      float mx[8];
#pragma unroll
      for (int i = 0; i < 8; ++i) mx[i] = -__builtin_inff();
#pragma unroll
      for (int nf = 0; nf < 4; ++nf)
#pragma unroll
        for (int i = 0; i < 8; ++i) {
          float vv = s[nf][i] * 0.03125f;               // E^-0.5
          if (domask) {
            int key = k0 + nf * 16 + (L & 15);
            int qq = qw0 + ((L >> 4) << 3) + i;
            if (key > qq) vv = -__builtin_inff();
          }
          s[nf][i] = vv;
          mx[i] = fmaxf(mx[i], vv);
        }
#pragma unroll
      for (int i = 0; i < 8; ++i)
#pragma unroll
        for (int off = 1; off < 16; off <<= 1)
          mx[i] = fmaxf(mx[i], __shfl_xor(mx[i], off));

      float corr[8];
#pragma unroll
      for (int i = 0; i < 8; ++i) {
        float nm = fmaxf(ml[i], mx[i]);
        corr[i] = __expf(ml[i] - nm);
        ml[i] = nm;
      }
      float pr[4][8];
#pragma unroll
      for (int nf = 0; nf < 4; ++nf)
#pragma unroll
        for (int i = 0; i < 8; ++i) pr[nf][i] = __expf(s[nf][i] - ml[i]);
#pragma unroll
      for (int i = 0; i < 8; ++i) {
        float r = pr[0][i] + pr[1][i] + pr[2][i] + pr[3][i];
#pragma unroll
        for (int off = 1; off < 16; off <<= 1) r += __shfl_xor(r, off);
        ll[i] = ll[i] * corr[i] + r;
      }
#pragma unroll
      for (int df = 0; df < 4; ++df)
#pragma unroll
        for (int i = 0; i < 8; ++i) o[df][i] *= corr[i];

      // relayout P (C/D layout -> A layout) through per-wave LDS
#pragma unroll
      for (int nf = 0; nf < 4; ++nf)
#pragma unroll
        for (int i = 0; i < 8; ++i)
          ldsP[w][(((L >> 4) << 3) + i) * 72 + nf * 16 + (L & 15)] = f2bf(pr[nf][i]);
      FragU pa0, pa1;
      {
        int prow = (L & 15);
        int kblk = (L >> 4) * 8;
        pa0.q[0] = *(const u32x4*)&ldsP[w][prow * 72 + kblk];
        pa0.q[1] = *(const u32x4*)&ldsP[w][prow * 72 + kblk + 16];
        pa1.q[0] = *(const u32x4*)&ldsP[w][prow * 72 + kblk + 32];
        pa1.q[1] = *(const u32x4*)&ldsP[w][prow * 72 + kblk + 48];
      }
      // O += P V (4 D-frags x 2 K-steps)
#pragma unroll
      for (int df = 0; df < 4; ++df) {
        FragU vf0, vf1;
        int dd = df * 16 + (L & 15);
        int kb2 = (L >> 4) * 8;
        vf0.q[0] = *(const u32x4*)&lv[dd * 72 + kb2];
        vf0.q[1] = *(const u32x4*)&lv[dd * 72 + kb2 + 16];
        vf1.q[0] = *(const u32x4*)&lv[dd * 72 + 32 + kb2];
        vf1.q[1] = *(const u32x4*)&lv[dd * 72 + 32 + kb2 + 16];
        o[df] = wmma_bf16(pa0, vf0, o[df]);
        o[df] = wmma_bf16(pa1, vf1, o[df]);
      }
    }
    p ^= 1;
  }

  // normalize and store attn output as bf16 [B*T, E]
#pragma unroll
  for (int df = 0; df < 4; ++df)
#pragma unroll
    for (int i = 0; i < 8; ++i) {
      int row = qw0 + ((L >> 4) << 3) + i;
      int d = df * 16 + (L & 15);
      float vv = o[df][i] / ll[i];
      attn[((size_t)b * 2048 + row) * 1024 + h * 64 + d] = f2bf(vv);
    }
}

// ---------------------------------------------------------------------------
// Host orchestration
// ---------------------------------------------------------------------------
extern "C" void kernel_launch(void* const* d_in, const int* in_sizes, int n_in,
                              void* d_out, int out_size, void* d_ws, size_t ws_size,
                              hipStream_t stream) {
  (void)in_sizes; (void)n_in; (void)out_size; (void)ws_size;
  const float* x   = (const float*)d_in[0];
  const float* Wq  = (const float*)d_in[1];
  const float* Wk  = (const float*)d_in[2];
  const float* Wv  = (const float*)d_in[3];
  const float* Wp  = (const float*)d_in[4];
  const float* bp  = (const float*)d_in[5];
  const float* W1  = (const float*)d_in[6];
  const float* b1  = (const float*)d_in[7];
  const float* W2  = (const float*)d_in[8];
  const float* b2  = (const float*)d_in[9];
  const float* g1  = (const float*)d_in[10];
  const float* be1 = (const float*)d_in[11];
  const float* g2  = (const float*)d_in[12];
  const float* be2 = (const float*)d_in[13];
  float* out = (float*)d_out;

  const size_t EE = 1024ull * 1024, EF = 1024ull * 4096;
  const size_t ME = 8192ull * 1024, MF = 8192ull * 4096;
  us* ws = (us*)d_ws;
  us* wqT = ws;        us* wkT = wqT + EE;  us* wvT = wkT + EE;  us* wpT = wvT + EE;
  us* w1T = wpT + EE;  us* w2T = w1T + EF;
  us* h1  = w2T + EF;  us* qb  = h1 + ME;   us* kb  = qb + ME;   us* vt  = kb + ME;
  us* at  = vt + ME;   us* h2  = at + ME;   us* ff  = h2 + ME;
  float* x1 = (float*)(ff + MF);

  auto cvtT = [&](const float* s, us* d, int K, int N) {
    k_cvtT<<<dim3(K / 32, N / 32), 256, 0, stream>>>(s, d, K, N);
  };
  cvtT(Wq, wqT, 1024, 1024); cvtT(Wk, wkT, 1024, 1024);
  cvtT(Wv, wvT, 1024, 1024); cvtT(Wp, wpT, 1024, 1024);
  cvtT(W1, w1T, 1024, 4096); cvtT(W2, w2T, 4096, 1024);

  k_layernorm<<<8192, 256, 0, stream>>>(x, g1, be1, h1);

  dim3 gEE(1024 / 128, 8192 / 128);
  k_gemm<0><<<gEE, 256, 0, stream>>>(h1, wqT, nullptr, nullptr, qb, nullptr, 8192, 1024, 1024);
  k_gemm<0><<<gEE, 256, 0, stream>>>(h1, wkT, nullptr, nullptr, kb, nullptr, 8192, 1024, 1024);
  k_gemm<3><<<gEE, 256, 0, stream>>>(h1, wvT, nullptr, nullptr, vt, nullptr, 8192, 1024, 1024);

  k_attention<<<dim3(32, 64), 128, 0, stream>>>(qb, kb, vt, at);

  k_gemm<1><<<gEE, 256, 0, stream>>>(at, wpT, bp, x, nullptr, x1, 8192, 1024, 1024);

  k_layernorm<<<8192, 256, 0, stream>>>(x1, g2, be2, h2);

  dim3 gEF(4096 / 128, 8192 / 128);
  k_gemm<2><<<gEF, 256, 0, stream>>>(h2, w1T, b1, nullptr, ff, nullptr, 8192, 4096, 1024);

  k_gemm<1><<<gEE, 256, 0, stream>>>(ff, w2T, b2, x1, nullptr, out, 8192, 1024, 4096);
}
